// COV_75359496176097
// MI455X (gfx1250) — matrix operand order repaired and checked by reading, not compiled, not run on hardware
//
#include <hip/hip_runtime.h>

typedef __attribute__((ext_vector_type(16))) __bf16 v16bf;
typedef __attribute__((ext_vector_type(8)))  __bf16 v8bf;
typedef __attribute__((ext_vector_type(4)))  __bf16 bf16x4;
typedef __attribute__((ext_vector_type(8)))  float  v8f;
typedef __attribute__((ext_vector_type(4)))  float  f32x4;

#define ZDIM   128
#define MDIM   8192
#define KDIM   8192
#define NBATCH 128
#define KC     32
#define NITER  (KDIM / KC)          // 256
#define LPITCH 48                   // 32 data + 16 pad bf16 elems => 96B rows, 16B aligned
#define PLANE  1048576u             // elements per eps plane (128 x 8192)

struct bf2 { __bf16 hi, lo; };
__device__ __forceinline__ bf2 bf16_split(float x) {
  bf2 r;
  r.hi = (__bf16)x;
  r.lo = (__bf16)(x - (float)r.hi);
  return r;
}

// ---- CDNA5 async global->LDS DMA (ASYNCcnt), two 16B pieces 16B apart ----
// inst offset is added to BOTH the LDS and global addresses (ISA 15.18.3).
__device__ __forceinline__ void async_pair_b128(unsigned lds, unsigned goff,
                                                unsigned long long sbase) {
  asm volatile("global_load_async_to_lds_b128 %0, %1, %2 offset:0"
               :: "v"(lds), "v"(goff), "s"(sbase) : "memory");
  asm volatile("global_load_async_to_lds_b128 %0, %1, %2 offset:16"
               :: "v"(lds), "v"(goff), "s"(sbase) : "memory");
}

__device__ __forceinline__ void wait_async0() {
#if __has_builtin(__builtin_amdgcn_s_wait_asynccnt)
  __builtin_amdgcn_s_wait_asynccnt(0);
#else
  asm volatile("s_wait_asynccnt 0" ::: "memory");
#endif
}

// ---------------- pre-pass: split eps into bf16 hi/lo planes in d_ws ----------------
__global__ __launch_bounds__(256)
void eps_split_kernel(const float* __restrict__ eps,
                      __bf16* __restrict__ hi, __bf16* __restrict__ lo)
{
  const unsigned i = (blockIdx.x * blockDim.x + threadIdx.x) * 4u;  // 4 elems/thread
  f32x4 v = *(const f32x4*)(eps + i);
  bf16x4 h, l;
  #pragma unroll
  for (int j = 0; j < 4; ++j) {
    bf2 s = bf16_split(v[j]);
    h[j] = s.hi; l[j] = s.lo;
  }
  *(bf16x4*)(hi + i) = h;
  *(bf16x4*)(lo + i) = l;
}

// ---------------- broadcast outputs: mu_out and logvar ----------------
__global__ __launch_bounds__(256)
void cov_broadcast_kernel(const float* __restrict__ mu,
                          const float* __restrict__ logstd,
                          float* __restrict__ out)
{
  const unsigned idx = blockIdx.x * blockDim.x + threadIdx.x;  // 0 .. 2^20
  const unsigned zi  = idx & (ZDIM - 1);
  __builtin_nontemporal_store(mu[zi], &out[idx]);
  __builtin_nontemporal_store(2.0f * logstd[zi], &out[1048576u + idx]);
}

// ---------------- fused rownorm + scaled GEMM (bf16x3 WMMA, async-DMA eps) ----------------
__global__ __launch_bounds__(256)
void cov_sample_kernel(const float* __restrict__ mu,
                       const float* __restrict__ logstd,
                       const float* __restrict__ Bm,
                       const __bf16* __restrict__ ehi,
                       const __bf16* __restrict__ elo,
                       float* __restrict__ out)
{
  __shared__ __align__(16) __bf16 sh_hi[2][NBATCH * LPITCH];  // 2 x 12 KB
  __shared__ __align__(16) __bf16 sh_lo[2][NBATCH * LPITCH];  // 2 x 12 KB

  const int tid   = threadIdx.x;
  const int wave  = tid >> 5;          // 8 waves
  const int lane  = tid & 31;          // wave32
  const int khalf = (lane >> 4) & 1;   // K half owned by this lane
  const int mloc  = lane & 15;         // B-row within the wave's 16-row strip
  const int m0    = blockIdx.x * 128 + wave * 16;
  const int mrow  = m0 + mloc;

  // DMA assignment: 256 threads cover 128 rows x 2 (16B-pair) pieces per plane
  const int srow = tid & 127;
  const int pg   = tid >> 7;
  const unsigned goff = (unsigned)(srow * KDIM + pg * 16) * 2u;  // bytes into a plane row
  const unsigned lh0 = (unsigned)(uintptr_t)&sh_hi[0][srow * LPITCH + pg * 16];
  const unsigned lh1 = (unsigned)(uintptr_t)&sh_hi[1][srow * LPITCH + pg * 16];
  const unsigned ll0 = (unsigned)(uintptr_t)&sh_lo[0][srow * LPITCH + pg * 16];
  const unsigned ll1 = (unsigned)(uintptr_t)&sh_lo[1][srow * LPITCH + pg * 16];
  const unsigned long long hbase = (unsigned long long)(uintptr_t)ehi;
  const unsigned long long lbase = (unsigned long long)(uintptr_t)elo;

  v8f   acc[8] = {};
  float ss = 0.0f;

  const float* browp = Bm + (size_t)mrow * KDIM + khalf * 8;

  // prologue: DMA chunk 0 into buffer 0
  async_pair_b128(lh0, goff, hbase);
  async_pair_b128(ll0, goff, lbase);

  for (int i = 0; i < NITER; ++i) {
    wait_async0();        // this wave's DMAs for chunk i have landed
    __syncthreads();      // all waves' DMAs landed; prev reads of 'next' buffer done

    if (i + 1 < NITER) {  // overlap chunk i+1 DMA with chunk i compute
      const unsigned long long koff = (unsigned long long)((i + 1) * KC) * 2ull;
      async_pair_b128((i & 1) ? lh0 : lh1, goff, hbase + koff);
      async_pair_b128((i & 1) ? ll0 : ll1, goff, lbase + koff);
    }

    // ---- this wave's 16 B rows: non-temporal stream, split, sumsq ----
    v16bf bhi, blo;
    {
      const f32x4* bp = (const f32x4*)(browp + i * KC);
      #pragma unroll
      for (int c = 0; c < 4; ++c) {
        f32x4 q = __builtin_nontemporal_load(bp + (c & 1) + (c >> 1) * 4);
        #pragma unroll
        for (int j = 0; j < 4; ++j) {
          const float f = q[j];
          ss = fmaf(f, f, ss);
          bf2 s = bf16_split(f);
          bhi[4 * c + j] = s.hi;
          blo[4 * c + j] = s.lo;
        }
      }
    }

    // ---- 8 batch tiles x 3 split-precision WMMAs ----
    const __bf16* bufh = sh_hi[i & 1];
    const __bf16* bufl = sh_lo[i & 1];
    #pragma unroll
    for (int t = 0; t < 8; ++t) {
      const int r = t * 16 + mloc;
      const __bf16* ph = &bufh[r * LPITCH + khalf * 8];
      const __bf16* pl = &bufl[r * LPITCH + khalf * 8];
      v8bf eh0 = *(const v8bf*)(ph);
      v8bf eh1 = *(const v8bf*)(ph + 16);
      v8bf el0 = *(const v8bf*)(pl);
      v8bf el1 = *(const v8bf*)(pl + 16);
      v16bf eah = __builtin_shufflevector(eh0, eh1, 0,1,2,3,4,5,6,7,8,9,10,11,12,13,14,15);
      v16bf eal = __builtin_shufflevector(el0, el1, 0,1,2,3,4,5,6,7,8,9,10,11,12,13,14,15);
      acc[t] = __builtin_amdgcn_wmma_f32_16x16x32_bf16(
          false, eah, false, bhi, (short)0, acc[t], false, false);
      acc[t] = __builtin_amdgcn_wmma_f32_16x16x32_bf16(
          false, eah, false, blo, (short)0, acc[t], false, false);
      acc[t] = __builtin_amdgcn_wmma_f32_16x16x32_bf16(
          false, eal, false, bhi, (short)0, acc[t], false, false);
    }
    __syncthreads();      // compute-i reads done before chunk i+2 overwrites
  }

  // ---- per-row scale: rsqrt(sumsq) * exp(logstd), then mu + scale*acc ----
  const float sstot = ss + __shfl_xor(ss, 16, 32);   // combine lane K halves
  const float scale = rsqrtf(sstot) * __expf(logstd[mrow & (ZDIM - 1)]);
  const float murep = mu[mrow & (ZDIM - 1)];

  float* samples = out + 2u * 1048576u;
  #pragma unroll
  for (int t = 0; t < 8; ++t) {
    #pragma unroll
    for (int r = 0; r < 8; ++r) {
      const int b = t * 16 + r + 8 * khalf;   // D-layout: M(=batch) = vgpr + 8*lanehalf
      __builtin_nontemporal_store(fmaf(scale, acc[t][r], murep),
                                  &samples[(size_t)b * MDIM + mrow]);
    }
  }
}

extern "C" void kernel_launch(void* const* d_in, const int* in_sizes, int n_in,
                              void* d_out, int out_size, void* d_ws, size_t ws_size,
                              hipStream_t stream) {
  const float* mu     = (const float*)d_in[0];  // (1,128)
  const float* logstd = (const float*)d_in[1];  // (128,)
  const float* Bm     = (const float*)d_in[2];  // (1,8192,8192)
  const float* eps    = (const float*)d_in[3];  // (128,8192,1)
  float* out = (float*)d_out;                   // [mu_out | logvar | samples]

  __bf16* ehi = (__bf16*)d_ws;                  // 2 MB
  __bf16* elo = ehi + PLANE;                    // 2 MB

  eps_split_kernel<<<1024, 256, 0, stream>>>(eps, ehi, elo);
  cov_broadcast_kernel<<<4096, 256, 0, stream>>>(mu, logstd, out);
  cov_sample_kernel<<<64, 256, 0, stream>>>(mu, logstd, Bm, ehi, elo, out);
}
